// CustomGNN_1_18975165513857
// MI455X (gfx1250) — compile-verified, matrix operand order checked by reading, not run on hardware
//
#include <hip/hip_runtime.h>
#include <stdint.h>

#define HIDDEN  128
#define NGRAPHS 512
#define BN_EPS  1e-5f

typedef __attribute__((ext_vector_type(16))) __bf16 v16bf;
typedef __attribute__((ext_vector_type(8)))  float  v8f;
typedef __attribute__((ext_vector_type(8)))  __bf16 v8bf;   // 16 bytes
typedef __attribute__((ext_vector_type(4)))  int    v4i;    // 16 bytes

union Frag16 {
    v16bf  v;
    v8bf   h[2];
    __bf16 e[16];
};

#if defined(__has_builtin)
#if __has_builtin(__builtin_amdgcn_global_load_async_to_lds_b128) && \
    __has_builtin(__builtin_amdgcn_s_wait_asynccnt)
#define HAVE_ASYNC_LDS 1
#endif
#endif

#define GLOBAL_AS __attribute__((address_space(1)))
#define LDS_AS    __attribute__((address_space(3)))

__device__ __forceinline__ v8f wmma_bf16(v16bf a, v16bf b, v8f c) {
    // D = A(16x32 bf16) x B(32x16 bf16) + C(16x16 f32)
    return __builtin_amdgcn_wmma_f32_16x16x32_bf16(false, a, false, b,
                                                   (short)0, c, false, false);
}

__device__ __forceinline__ v8f splat8(float x) {
    v8f r = {x, x, x, x, x, x, x, x};
    return r;
}

__device__ __forceinline__ void wave_sync_lds() {
    __builtin_amdgcn_wave_barrier();
    asm volatile("s_wait_dscnt 0" ::: "memory");
    __builtin_amdgcn_wave_barrier();
}

// Stage nChunks16B x 16 bytes from global to LDS (gfx1250 async-copy when available).
__device__ __forceinline__ void stage_w_lds(__bf16* lds, const __bf16* glob,
                                            int nChunks16B, int tid, int nthreads) {
#ifdef HAVE_ASYNC_LDS
    for (int i = tid; i < nChunks16B; i += nthreads)
        __builtin_amdgcn_global_load_async_to_lds_b128(
            (GLOBAL_AS v4i*)(glob + i * 8),
            (LDS_AS v4i*)(lds + i * 8), 0, 0);
    __builtin_amdgcn_s_wait_asynccnt(0);
#else
    for (int i = tid; i < nChunks16B; i += nthreads)
        *(v8bf*)(lds + i * 8) = *(const v8bf*)(glob + i * 8);
#endif
}

// ---------------------------------------------------------------- utilities

__global__ __launch_bounds__(256) void zero_f32_kernel(float* __restrict__ p, int n) {
    for (int i = blockIdx.x * blockDim.x + threadIdx.x; i < n; i += gridDim.x * blockDim.x)
        p[i] = 0.0f;
}

// ReLU f32 accumulator -> bf16 node-feature table (hardware cvt)
__global__ __launch_bounds__(256) void relu_cvt_kernel(const float* __restrict__ in,
                                                       __bf16* __restrict__ out, int n) {
    for (int i = blockIdx.x * blockDim.x + threadIdx.x; i < n; i += gridDim.x * blockDim.x)
        out[i] = (__bf16)fmaxf(in[i], 0.0f);
}

__global__ __launch_bounds__(256) void f32_to_bf16_kernel(const float* __restrict__ in,
                                                          __bf16* __restrict__ out, int n) {
    for (int i = blockIdx.x * blockDim.x + threadIdx.x; i < n; i += gridDim.x * blockDim.x)
        out[i] = (__bf16)in[i];
}

// Swizzle row-major W[K][128] (f32) into per-lane B-fragment layout (bf16):
// fragment (kt,nt): 32 lanes x 16 bf16 (32B) contiguous per lane.
// dword d of lane holds K = kt*32 + kb + 2*(d&3) + (d>=4)*16, kb = (lane>=16)*8,
// column N = nt*16 + (lane&15).
__global__ __launch_bounds__(256) void swizzle_w_kernel(const float* __restrict__ W,
                                                        __bf16* __restrict__ out,
                                                        int Ktiles, int Kreal) {
    int idx = blockIdx.x * blockDim.x + threadIdx.x;
    int total = Ktiles * 8 * 32;
    if (idx >= total) return;
    int lane = idx & 31;
    int nt   = (idx >> 5) & 7;
    int kt   = idx >> 8;
    int col  = nt * 16 + (lane & 15);
    int kb   = (lane >> 4) * 8;
    __bf16* o = out + (size_t)idx * 16;
#pragma unroll
    for (int d = 0; d < 8; ++d) {
        int k = kt * 32 + kb + 2 * (d & 3) + ((d >> 2) * 16);
        o[2 * d]     = (k     < Kreal) ? (__bf16)W[(size_t)k       * HIDDEN + col] : (__bf16)0.0f;
        o[2 * d + 1] = (k + 1 < Kreal) ? (__bf16)W[(size_t)(k + 1) * HIDDEN + col] : (__bf16)0.0f;
    }
}

// ---------------------------------------------------------------- conv kernels

// One wave = one 16-edge tile. lin1 -> ReLU -> LDS restage -> lin2 -> atomic scatter.
// All weights LDS-resident (async-staged); A gather is pure b128 loads, issued as one
// clause per tile so there is a single load-wait before 72 back-to-back WMMAs.
__global__ __launch_bounds__(128, 2) void conv2_kernel(
        const __bf16* __restrict__ hin, const float* __restrict__ pos,
        const int* __restrict__ src, const int* __restrict__ dst,
        const __bf16* __restrict__ w1s, const float* __restrict__ b1,
        const __bf16* __restrict__ w2s, const float* __restrict__ b2,
        float* __restrict__ out, int numTiles, int E) {
    __shared__ __align__(32) __bf16 w1l[9 * 8 * 32 * 16];    // 72KB lin1 weights
    __shared__ __align__(32) __bf16 w2l[4 * 8 * 32 * 16];    // 32KB lin2 weights
    __shared__ __align__(32) __bf16 h1t[4][16 * HIDDEN];     // 16KB D->A restage

    stage_w_lds(w1l, w1s, 9 * 8 * 32 * 2, threadIdx.x, 128);
    stage_w_lds(w2l, w2s, 4 * 8 * 32 * 2, threadIdx.x, 128);
    __syncthreads();

    const int wave = threadIdx.x >> 5, lane = threadIdx.x & 31;
    const int row = lane & 15, hi = lane >> 4;
    const int kb = hi * 8;
    __bf16* t = h1t[wave];

    float bias1[8], bias2[8];
#pragma unroll
    for (int nt = 0; nt < 8; ++nt) {
        bias1[nt] = b1[nt * 16 + row];
        bias2[nt] = b2[nt * 16 + row];
    }

    const int tileStep = gridDim.x * 4;
    for (int tile = blockIdx.x * 4 + wave; tile < numTiles; tile += tileStep) {
        const int ebase = tile * 16;
        // prefetch next tile's edge indices
        __builtin_prefetch((const void*)(src + ebase + tileStep * 16), 0, 0);
        __builtin_prefetch((const void*)(dst + ebase + tileStep * 16), 0, 0);

        int e = ebase + row; if (e >= E) e = E - 1;
        const int sI = src[e], dI = dst[e];
        // scatter indices loaded early so they overlap the matrix math
        int dstm[8];
#pragma unroll
        for (int r = 0; r < 8; ++r) {
            int em = ebase + r + hi * 8;
            dstm[r] = (em < E) ? dst[em] : -1;
        }

        const __bf16* hD = hin + (size_t)dI * HIDDEN;
        const __bf16* hS = hin + (size_t)sI * HIDDEN;
        const float dp0 = pos[sI * 3 + 0] - pos[dI * 3 + 0];
        const float dp1 = pos[sI * 3 + 1] - pos[dI * 3 + 1];
        const float dp2 = pos[sI * 3 + 2] - pos[dI * 3 + 2];

        // ---- load the full 16x288 A tile (9 fragments) up front: one clause, one wait
        Frag16 a[9];
#pragma unroll
        for (int kt = 0; kt < 4; ++kt) {               // K 0..127 : h[dst]
            a[kt].h[0] = *(const v8bf*)(hD + kt * 32 + kb);
            a[kt].h[1] = *(const v8bf*)(hD + kt * 32 + kb + 16);
        }
#pragma unroll
        for (int kt = 0; kt < 4; ++kt) {               // K 128..255 : h[src]
            a[4 + kt].h[0] = *(const v8bf*)(hS + kt * 32 + kb);
            a[4 + kt].h[1] = *(const v8bf*)(hS + kt * 32 + kb + 16);
        }
#pragma unroll
        for (int j = 0; j < 16; ++j) {                 // K 256..287 : dpos + pad
            int k = 256 + kb + (j & 7) + ((j >> 3) * 16);
            float v = (k == 256) ? dp0 : (k == 257) ? dp1 : (k == 258) ? dp2 : 0.0f;
            a[8].e[j] = (__bf16)v;
        }

        // ---- lin1: [16x288] x [288x128], bias in C init, weights from LDS
        v8f acc[8];
#pragma unroll
        for (int nt = 0; nt < 8; ++nt) acc[nt] = splat8(bias1[nt]);
#pragma unroll
        for (int kt = 0; kt < 9; ++kt)
#pragma unroll
            for (int nt = 0; nt < 8; ++nt) {
                const Frag16* bp = reinterpret_cast<const Frag16*>(w1l) + ((kt * 8 + nt) * 32 + lane);
                acc[nt] = wmma_bf16(a[kt].v, bp->v, acc[nt]);
            }

        // ---- ReLU, D(lane=col) -> A(lane=row) restage through LDS as bf16
#pragma unroll
        for (int nt = 0; nt < 8; ++nt)
#pragma unroll
            for (int r = 0; r < 8; ++r)
                t[(r + hi * 8) * HIDDEN + nt * 16 + row] = (__bf16)fmaxf(acc[nt][r], 0.0f);
        wave_sync_lds();

        // ---- lin2: [16x128] x [128x128], weights from LDS
#pragma unroll
        for (int nt = 0; nt < 8; ++nt) acc[nt] = splat8(bias2[nt]);
#pragma unroll
        for (int kt = 0; kt < 4; ++kt) {
            Frag16 a2;
            a2.h[0] = *(const v8bf*)(t + row * HIDDEN + kt * 32 + kb);
            a2.h[1] = *(const v8bf*)(t + row * HIDDEN + kt * 32 + kb + 16);
#pragma unroll
            for (int nt = 0; nt < 8; ++nt) {
                const Frag16* bp = reinterpret_cast<const Frag16*>(w2l) + ((kt * 8 + nt) * 32 + lane);
                acc[nt] = wmma_bf16(a2.v, bp->v, acc[nt]);
            }
        }
        wave_sync_lds();   // protect h1t WAR for next tile

        // ---- scatter-add into destination nodes
#pragma unroll
        for (int nt = 0; nt < 8; ++nt)
#pragma unroll
            for (int r = 0; r < 8; ++r)
                if (dstm[r] >= 0)
                    atomicAdd(&out[(size_t)dstm[r] * HIDDEN + nt * 16 + row], acc[nt][r]);
    }
}

__global__ __launch_bounds__(128, 2) void conv1_kernel(
        const float* __restrict__ x, const float* __restrict__ pos,
        const int* __restrict__ src, const int* __restrict__ dst,
        const __bf16* __restrict__ w1s, const float* __restrict__ b1,
        const __bf16* __restrict__ w2s, const float* __restrict__ b2,
        float* __restrict__ out, int numTiles, int E) {
    __shared__ __align__(32) __bf16 w1l[1 * 8 * 32 * 16];    // 8KB
    __shared__ __align__(32) __bf16 w2l[4 * 8 * 32 * 16];    // 32KB
    __shared__ __align__(32) __bf16 h1t[4][16 * HIDDEN];     // 16KB

    stage_w_lds(w1l, w1s, 1 * 8 * 32 * 2, threadIdx.x, 128);
    stage_w_lds(w2l, w2s, 4 * 8 * 32 * 2, threadIdx.x, 128);
    __syncthreads();

    const int wave = threadIdx.x >> 5, lane = threadIdx.x & 31;
    const int row = lane & 15, hi = lane >> 4;
    const int kb = hi * 8;
    __bf16* t = h1t[wave];

    float bias1[8], bias2[8];
#pragma unroll
    for (int nt = 0; nt < 8; ++nt) {
        bias1[nt] = b1[nt * 16 + row];
        bias2[nt] = b2[nt * 16 + row];
    }

    const int tileStep = gridDim.x * 4;
    for (int tile = blockIdx.x * 4 + wave; tile < numTiles; tile += tileStep) {
        const int ebase = tile * 16;
        __builtin_prefetch((const void*)(src + ebase + tileStep * 16), 0, 0);
        __builtin_prefetch((const void*)(dst + ebase + tileStep * 16), 0, 0);

        int e = ebase + row; if (e >= E) e = E - 1;
        const int sI = src[e], dI = dst[e];
        int dstm[8];
#pragma unroll
        for (int r = 0; r < 8; ++r) {
            int em = ebase + r + hi * 8;
            dstm[r] = (em < E) ? dst[em] : -1;
        }

        const float f0 = x[dI], f1 = x[sI];
        const float dp0 = pos[sI * 3 + 0] - pos[dI * 3 + 0];
        const float dp1 = pos[sI * 3 + 1] - pos[dI * 3 + 1];
        const float dp2 = pos[sI * 3 + 2] - pos[dI * 3 + 2];

        // ---- lin1: feat row = [x_dst, x_src, dpos(3), 0...] (K padded to 32)
        Frag16 a;
#pragma unroll
        for (int j = 0; j < 16; ++j) {
            int k = kb + (j & 7) + ((j >> 3) * 16);
            float v = (k == 0) ? f0 : (k == 1) ? f1
                    : (k == 2) ? dp0 : (k == 3) ? dp1 : (k == 4) ? dp2 : 0.0f;
            a.e[j] = (__bf16)v;
        }
        v8f acc[8];
#pragma unroll
        for (int nt = 0; nt < 8; ++nt) {
            acc[nt] = splat8(bias1[nt]);
            const Frag16* bp = reinterpret_cast<const Frag16*>(w1l) + (nt * 32 + lane);
            acc[nt] = wmma_bf16(a.v, bp->v, acc[nt]);
        }

#pragma unroll
        for (int nt = 0; nt < 8; ++nt)
#pragma unroll
            for (int r = 0; r < 8; ++r)
                t[(r + hi * 8) * HIDDEN + nt * 16 + row] = (__bf16)fmaxf(acc[nt][r], 0.0f);
        wave_sync_lds();

#pragma unroll
        for (int nt = 0; nt < 8; ++nt) acc[nt] = splat8(bias2[nt]);
#pragma unroll
        for (int kt = 0; kt < 4; ++kt) {
            Frag16 a2;
            a2.h[0] = *(const v8bf*)(t + row * HIDDEN + kt * 32 + kb);
            a2.h[1] = *(const v8bf*)(t + row * HIDDEN + kt * 32 + kb + 16);
#pragma unroll
            for (int nt = 0; nt < 8; ++nt) {
                const Frag16* bp = reinterpret_cast<const Frag16*>(w2l) + ((kt * 8 + nt) * 32 + lane);
                acc[nt] = wmma_bf16(a2.v, bp->v, acc[nt]);
            }
        }
        wave_sync_lds();

#pragma unroll
        for (int nt = 0; nt < 8; ++nt)
#pragma unroll
            for (int r = 0; r < 8; ++r)
                if (dstm[r] >= 0)
                    atomicAdd(&out[(size_t)dstm[r] * HIDDEN + nt * 16 + row], acc[nt][r]);
    }
}

// ---------------------------------------------------------------- pooling + predictor

__global__ __launch_bounds__(256) void pool_relu_kernel(const float* __restrict__ h2,
                                                        const int* __restrict__ batch,
                                                        float* __restrict__ g, int n) {
    for (int i = blockIdx.x * blockDim.x + threadIdx.x; i < n; i += gridDim.x * blockDim.x) {
        int v = i >> 7, c = i & 127;
        float val = fmaxf(h2[i], 0.0f);
        atomicAdd(&g[(size_t)batch[v] * HIDDEN + c], val);
    }
}

// z = g(512x128) @ p_w1(128x128); one wave per 16-row M tile.
__global__ __launch_bounds__(128) void pred_gemm_kernel(const __bf16* __restrict__ gbf,
                                                        const __bf16* __restrict__ w1s,
                                                        float* __restrict__ z) {
    const int wave = threadIdx.x >> 5, lane = threadIdx.x & 31;
    const int row = lane & 15, hi = lane >> 4;
    const int kb = hi * 8;
    const int mt = blockIdx.x * 4 + wave;
    if (mt >= NGRAPHS / 16) return;

    v8f acc[8];
#pragma unroll
    for (int nt = 0; nt < 8; ++nt) acc[nt] = splat8(0.0f);
#pragma unroll
    for (int kt = 0; kt < 4; ++kt) {
        Frag16 a;
        const __bf16* ap = gbf + (size_t)(mt * 16 + row) * HIDDEN + kt * 32 + kb;
        a.h[0] = *(const v8bf*)ap;
        a.h[1] = *(const v8bf*)(ap + 16);
#pragma unroll
        for (int nt = 0; nt < 8; ++nt) {
            const Frag16* bp = reinterpret_cast<const Frag16*>(w1s) + ((kt * 8 + nt) * 32 + lane);
            acc[nt] = wmma_bf16(a.v, bp->v, acc[nt]);
        }
    }
#pragma unroll
    for (int nt = 0; nt < 8; ++nt)
#pragma unroll
        for (int r = 0; r < 8; ++r)
            z[(size_t)(mt * 16 + r + hi * 8) * HIDDEN + nt * 16 + row] = acc[nt][r];
}

__global__ __launch_bounds__(128) void bn_stats_kernel(const float* __restrict__ z,
                                                       float* __restrict__ st) {
    int c = threadIdx.x;
    float s = 0.0f, q = 0.0f;
    for (int r = 0; r < NGRAPHS; ++r) {
        float v = z[(size_t)r * HIDDEN + c];
        s += v; q += v * v;
    }
    float mean = s * (1.0f / NGRAPHS);
    float var  = q * (1.0f / NGRAPHS) - mean * mean;
    st[c]          = mean;
    st[HIDDEN + c] = __frsqrt_rn(var + BN_EPS);
}

__global__ __launch_bounds__(256) void pred_out_kernel(const float* __restrict__ z,
                                                       const float* __restrict__ st,
                                                       const float* __restrict__ gamma,
                                                       const float* __restrict__ beta,
                                                       const float* __restrict__ pw2,
                                                       const float* __restrict__ pb2,
                                                       float* __restrict__ out) {
    int r = blockIdx.x * blockDim.x + threadIdx.x;
    if (r >= NGRAPHS) return;
    float s = 0.0f;
    for (int c = 0; c < HIDDEN; ++c) {
        float v = z[(size_t)r * HIDDEN + c];
        v = gamma[c] * (v - st[c]) * st[HIDDEN + c] + beta[c];
        s += fmaxf(v, 0.0f) * pw2[c];
    }
    out[r] = s + pb2[0];
}

// ---------------------------------------------------------------- launch

extern "C" void kernel_launch(void* const* d_in, const int* in_sizes, int n_in,
                              void* d_out, int out_size, void* d_ws, size_t ws_size,
                              hipStream_t stream) {
    const float* x     = (const float*)d_in[0];
    const float* pos   = (const float*)d_in[1];
    const int*   ei    = (const int*)d_in[2];
    const int*   batch = (const int*)d_in[3];
    const float* c1w1  = (const float*)d_in[4];
    const float* c1b1  = (const float*)d_in[5];
    const float* c1w2  = (const float*)d_in[6];
    const float* c1b2  = (const float*)d_in[7];
    const float* c2w1  = (const float*)d_in[8];
    const float* c2b1  = (const float*)d_in[9];
    const float* c2w2  = (const float*)d_in[10];
    const float* c2b2  = (const float*)d_in[11];
    const float* pw1   = (const float*)d_in[12];
    const float* gamma = (const float*)d_in[13];
    const float* beta  = (const float*)d_in[14];
    const float* pw2   = (const float*)d_in[15];
    const float* pb2   = (const float*)d_in[16];
    (void)n_in; (void)out_size; (void)ws_size;

    const int N = in_sizes[0];          // 50000 nodes
    const int E = in_sizes[2] / 2;      // 800000 edges
    const int* src = ei;
    const int* dst = ei + E;

    // workspace layout (256B aligned; h1/h2/g first so one zero pass covers all)
    char* ws = (char*)d_ws;
    size_t off = 0;
    auto take = [&](size_t bytes) { size_t o = off; off += (bytes + 255) & ~(size_t)255; return o; };
    float*  h1   = (float*)(ws + take((size_t)N * HIDDEN * 4));
    float*  h2   = (float*)(ws + take((size_t)N * HIDDEN * 4));
    float*  g    = (float*)(ws + take((size_t)NGRAPHS * HIDDEN * 4));
    __bf16* h1bf = (__bf16*)(ws + take((size_t)N * HIDDEN * 2));
    __bf16* gbf  = (__bf16*)(ws + take((size_t)NGRAPHS * HIDDEN * 2));
    float*  z    = (float*)(ws + take((size_t)NGRAPHS * HIDDEN * 4));
    float*  st   = (float*)(ws + take(2 * HIDDEN * 4));
    __bf16* w11s = (__bf16*)(ws + take((size_t)1 * 8 * 32 * 32));
    __bf16* w12s = (__bf16*)(ws + take((size_t)4 * 8 * 32 * 32));
    __bf16* w21s = (__bf16*)(ws + take((size_t)9 * 8 * 32 * 32));
    __bf16* w22s = (__bf16*)(ws + take((size_t)4 * 8 * 32 * 32));
    __bf16* pw1s = (__bf16*)(ws + take((size_t)4 * 8 * 32 * 32));

    // 1) zero accumulators (h1, h2, g are contiguous)
    {
        int nz = N * HIDDEN * 2 + NGRAPHS * HIDDEN;
        zero_f32_kernel<<<1024, 256, 0, stream>>>(h1, nz);
    }
    // 2) swizzle weights into B-fragment layout (bf16)
    swizzle_w_kernel<<<1, 256, 0, stream>>>(c1w1, w11s, 1, 5);
    swizzle_w_kernel<<<4, 256, 0, stream>>>(c1w2, w12s, 4, HIDDEN);
    swizzle_w_kernel<<<9, 256, 0, stream>>>(c2w1, w21s, 9, 2 * HIDDEN + 3);
    swizzle_w_kernel<<<4, 256, 0, stream>>>(c2w2, w22s, 4, HIDDEN);
    swizzle_w_kernel<<<4, 256, 0, stream>>>(pw1,  pw1s, 4, HIDDEN);

    const int numTiles = (E + 15) / 16;
    const int convBlocks = (numTiles + 3) / 4;

    // 3) conv1 -> h1 (raw f32 sums) -> ReLU + convert to bf16 feature table
    conv1_kernel<<<convBlocks, 128, 0, stream>>>(x, pos, src, dst, w11s, c1b1, w12s, c1b2,
                                                 h1, numTiles, E);
    relu_cvt_kernel<<<1024, 256, 0, stream>>>(h1, h1bf, N * HIDDEN);

    // 4) conv2 (bf16 gathers) -> h2 (raw f32 sums)
    conv2_kernel<<<convBlocks, 128, 0, stream>>>(h1bf, pos, src, dst, w21s, c2b1, w22s, c2b2,
                                                 h2, numTiles, E);

    // 5) ReLU + global add pool -> g
    pool_relu_kernel<<<1024, 256, 0, stream>>>(h2, batch, g, N * HIDDEN);

    // 6) predictor
    f32_to_bf16_kernel<<<(NGRAPHS * HIDDEN + 255) / 256, 256, 0, stream>>>(g, gbf, NGRAPHS * HIDDEN);
    pred_gemm_kernel<<<(NGRAPHS / 16 + 3) / 4, 128, 0, stream>>>(gbf, pw1s, z);
    bn_stats_kernel<<<1, 128, 0, stream>>>(z, st);
    pred_out_kernel<<<(NGRAPHS + 255) / 256, 256, 0, stream>>>(z, st, gamma, beta, pw2, pb2,
                                                               (float*)d_out);
}
